// AggregationLoss_78271484003260
// MI455X (gfx1250) — matrix operand order repaired and checked by reading, not compiled, not run on hardware
//
#include <hip/hip_runtime.h>
#include <math.h>

// AggregationLoss for MI455X (gfx1250, wave32).
// Segment sums implemented as one-hot WMMA matmuls:
//   D[16 inst x 16 stat] += onehot(id)[16 x 32] * stats[32 x 16]
// using V_WMMA_F32_16X16X32_F16 (one-hot exact in f16, f32 accumulate).
// All reductions are fixed-order (no atomics) => deterministic.

#define MAX_T  16
#define P      65536       // 256*256 pixels per sample
#define NSAMP  32
#define SLICES 16          // blocks per sample
#define WAVES  8
#define THREADS 256
#define PIX_PER_BLOCK (P / SLICES)            // 4096
#define PIX_PER_WAVE  (PIX_PER_BLOCK / WAVES) // 512

typedef __attribute__((ext_vector_type(16))) _Float16 v16h;
typedef __attribute__((ext_vector_type(8)))  float    v8f;

// Compiler-level ordering fence for wave-private LDS staging
// (HW keeps per-wave DS ops in order; this stops IR reordering).
__device__ __forceinline__ void lds_fence() { asm volatile("" ::: "memory"); }

// ---------------------------------------------------------------------------
// Pass 1: per (sample, slice) block, compute partial
//   stats[m][0..3] = sum_{kernel==m} sim_c   (WMMA, A=onehot(kernel))
//   stats[m][4]    = count(kernel==m)        (ones column of B)
//   stats[m][5]    = count(text==m)          (WMMA, A=onehot(text), col 4)
// VGPR layouts (CDNA5 ISA 7.12.2):
//   A 16x32 f16 : lane l (hsel=l>>4, n=l&15): elem e -> K = (e<8?e:e+8)+8*hsel, M=n
//   B 32x16 f16 : elem e -> K = (e<8?e:e+8)+8*hsel, N = n
//   D 16x16 f32 : vgpr r -> M = r + 8*hsel,  N = n
// ---------------------------------------------------------------------------
__global__ __launch_bounds__(THREADS) void agg_pass1(
    const float* __restrict__ preds, const int* __restrict__ targets,
    float* __restrict__ partial1) {
  __shared__ float simT[WAVES][4][32];   // channel-major staged sim
  __shared__ int   idsS[WAVES][32];      // kernel | text<<16
  __shared__ float red[WAVES][MAX_T][6];

  const int s    = blockIdx.x;
  const int b    = blockIdx.y;
  const int tid  = threadIdx.x;
  const int lane = tid & 31;
  const int wave = tid >> 5;
  const int hsel = lane >> 4;
  const int n    = lane & 15;

  const float* sim   = preds   + ((size_t)b * 6 + 2) * P;  // channels 2..5
  const int*   textp = targets + ((size_t)b * 2 + 0) * P;
  const int*   kernp = targets + ((size_t)b * 2 + 1) * P;

  v8f acc1 = {};  // onehot(kernel) x [sim0..3, 1, 0...]
  v8f acc2 = {};  // onehot(text)   x same B (only col 4 = t_cnt used)

  const int wbase = s * PIX_PER_BLOCK + wave * PIX_PER_WAVE;
  for (int c0 = 0; c0 < PIX_PER_WAVE; c0 += 32) {
    const int p = wbase + c0 + lane;
    const float s0 = sim[p];
    const float s1 = sim[P + p];
    const float s2 = sim[2 * P + p];
    const float s3 = sim[3 * P + p];
    const int   kt = kernp[p] | (textp[p] << 16);

    simT[wave][0][lane] = s0;
    simT[wave][1][lane] = s1;
    simT[wave][2][lane] = s2;
    simT[wave][3][lane] = s3;
    idsS[wave][lane]    = kt;
    lds_fence();

    v16h A1, A2, B;
#pragma unroll
    for (int e = 0; e < 16; ++e) {
      const int j   = ((e < 8) ? e : (e + 8)) + hsel * 8;   // A-layout K index
      const int id2 = idsS[wave][j];
      A1[e] = ((id2 & 0xffff) == n) ? (_Float16)1.0f : (_Float16)0.0f;
      A2[e] = ((id2 >> 16)    == n) ? (_Float16)1.0f : (_Float16)0.0f;
    }
    if (n < 4) {
#pragma unroll
      for (int e = 0; e < 16; ++e)
        B[e] = (_Float16)simT[wave][n][((e < 8) ? e : (e + 8)) + hsel * 8];      // B-layout K index
    } else if (n == 4) {
#pragma unroll
      for (int e = 0; e < 16; ++e) B[e] = (_Float16)1.0f;   // counts column
    } else {
#pragma unroll
      for (int e = 0; e < 16; ++e) B[e] = (_Float16)0.0f;
    }
    lds_fence();
    acc1 = __builtin_amdgcn_wmma_f32_16x16x32_f16(false, A1, false, B,
                                                  (short)0, acc1, false, false);
    acc2 = __builtin_amdgcn_wmma_f32_16x16x32_f16(false, A2, false, B,
                                                  (short)0, acc2, false, false);
  }

  // wave accumulators -> LDS (only stat columns 0..4 of acc1, col 4 of acc2)
  if (n < 5) {
#pragma unroll
    for (int r = 0; r < 8; ++r) red[wave][r + 8 * hsel][n] = acc1[r];
  }
  if (n == 4) {
#pragma unroll
    for (int r = 0; r < 8; ++r) red[wave][r + 8 * hsel][5] = acc2[r];
  }
  __syncthreads();

  if (tid < MAX_T * 6) {   // fixed-order cross-wave reduction
    const int m = tid / 6, st = tid % 6;
    float a = 0.f;
#pragma unroll
    for (int w = 0; w < WAVES; ++w) a += red[w][m][st];
    partial1[(((size_t)b * SLICES + s) * MAX_T + m) * 6 + st] = a;
  }
}

// ---------------------------------------------------------------------------
// Pass 2: reduce partial1 -> G, per-pixel loss, segment-sum by text via WMMA.
// ---------------------------------------------------------------------------
__global__ __launch_bounds__(THREADS) void agg_pass2(
    const float* __restrict__ preds, const int* __restrict__ targets,
    const float* __restrict__ partial1, float* __restrict__ partial2) {
  __shared__ float stats[MAX_T][6];
  __shared__ float GL[MAX_T][4];
  __shared__ float lpS[WAVES][32];
  __shared__ int   txS[WAVES][32];
  __shared__ float red[WAVES][MAX_T];

  const int s    = blockIdx.x;
  const int b    = blockIdx.y;
  const int tid  = threadIdx.x;
  const int lane = tid & 31;
  const int wave = tid >> 5;
  const int hsel = lane >> 4;
  const int n    = lane & 15;

  if (tid < MAX_T * 6) {
    const int m = tid / 6, st = tid % 6;
    float a = 0.f;
    for (int ss = 0; ss < SLICES; ++ss)
      a += partial1[(((size_t)b * SLICES + ss) * MAX_T + m) * 6 + st];
    stats[m][st] = a;
  }
  __syncthreads();
  if (tid < MAX_T * 4) {
    const int m = tid >> 2, c = tid & 3;
    GL[m][c] = stats[m][c] / fmaxf(stats[m][4], 1.0f);   // G = k_sum/max(k_cnt,1)
  }
  __syncthreads();

  const float* sim   = preds   + ((size_t)b * 6 + 2) * P;
  const int*   textp = targets + ((size_t)b * 2 + 0) * P;

  v8f acc = {};
  const int wbase = s * PIX_PER_BLOCK + wave * PIX_PER_WAVE;
  for (int c0 = 0; c0 < PIX_PER_WAVE; c0 += 32) {
    const int p  = wbase + c0 + lane;
    const int tx = textp[p];
    const float d0 = sim[p]         - GL[tx][0];
    const float d1 = sim[P + p]     - GL[tx][1];
    const float d2 = sim[2 * P + p] - GL[tx][2];
    const float d3 = sim[3 * P + p] - GL[tx][3];
    float d = sqrtf(d0 * d0 + d1 * d1 + d2 * d2 + d3 * d3) - 0.5f;
    d = fmaxf(d, 0.0f);
    const float lp = logf(d * d + 1.0f);

    lpS[wave][lane] = lp;
    txS[wave][lane] = tx;
    lds_fence();

    v16h A, B;
#pragma unroll
    for (int e = 0; e < 16; ++e) {
      const int j = ((e < 8) ? e : (e + 8)) + hsel * 8;
      A[e] = (txS[wave][j] == n) ? (_Float16)1.0f : (_Float16)0.0f;
    }
    if (n == 0) {
#pragma unroll
      for (int e = 0; e < 16; ++e) B[e] = (_Float16)lpS[wave][((e < 8) ? e : (e + 8)) + hsel * 8];
    } else {
#pragma unroll
      for (int e = 0; e < 16; ++e) B[e] = (_Float16)0.0f;
    }
    lds_fence();
    acc = __builtin_amdgcn_wmma_f32_16x16x32_f16(false, A, false, B,
                                                 (short)0, acc, false, false);
  }

  if (n == 0) {
#pragma unroll
    for (int r = 0; r < 8; ++r) red[wave][r + 8 * hsel] = acc[r];
  }
  __syncthreads();
  if (tid < MAX_T) {
    float a = 0.f;
#pragma unroll
    for (int w = 0; w < WAVES; ++w) a += red[w][tid];
    partial2[((size_t)b * SLICES + s) * MAX_T + tid] = a;
  }
}

// ---------------------------------------------------------------------------
// Final combine per sample.
// ---------------------------------------------------------------------------
__global__ __launch_bounds__(128) void agg_final(
    const float* __restrict__ partial1, const float* __restrict__ partial2,
    float* __restrict__ out) {
  __shared__ float stats[MAX_T][6];
  __shared__ float isum[MAX_T];
  const int b = blockIdx.x, tid = threadIdx.x;

  if (tid < MAX_T * 6) {
    const int m = tid / 6, st = tid % 6;
    float a = 0.f;
    for (int ss = 0; ss < SLICES; ++ss)
      a += partial1[(((size_t)b * SLICES + ss) * MAX_T + m) * 6 + st];
    stats[m][st] = a;
  }
  if (tid >= 96 && tid < 96 + MAX_T) {
    const int m = tid - 96;
    float a = 0.f;
    for (int ss = 0; ss < SLICES; ++ss)
      a += partial2[((size_t)b * SLICES + ss) * MAX_T + m];
    isum[m] = a;
  }
  __syncthreads();
  if (tid == 0) {
    float tot = 0.f, nv = 0.f;
    for (int m = 1; m < MAX_T; ++m) {         // skip background id 0
      const float kc = stats[m][4], tc = stats[m][5];
      if (kc > 0.f && tc > 0.f) {
        tot += isum[m] / fmaxf(tc, 1.f);
        nv  += 1.f;
      }
    }
    out[b] = (nv > 0.f) ? (tot / nv) : 0.f;
  }
}

extern "C" void kernel_launch(void* const* d_in, const int* in_sizes, int n_in,
                              void* d_out, int out_size, void* d_ws, size_t ws_size,
                              hipStream_t stream) {
  (void)in_sizes; (void)n_in; (void)out_size; (void)ws_size;
  const float* preds   = (const float*)d_in[0];   // [32,6,256,256] f32
  const int*   targets = (const int*)d_in[1];     // [32,2,256,256] i32
  float* out = (float*)d_out;                     // [32] f32
  float* ws  = (float*)d_ws;

  float* partial1 = ws;                                           // 32*16*16*6 f32
  float* partial2 = ws + (size_t)NSAMP * SLICES * MAX_T * 6;      // 32*16*16   f32

  dim3 grid(SLICES, NSAMP);
  agg_pass1<<<grid, THREADS, 0, stream>>>(preds, targets, partial1);
  agg_pass2<<<grid, THREADS, 0, stream>>>(preds, targets, partial1, partial2);
  agg_final<<<NSAMP, 128, 0, stream>>>(partial1, partial2, out);
}
